// SelfAttention_2637109920395
// MI455X (gfx1250) — matrix operand order repaired and checked by reading, not compiled, not run on hardware
//
#include <hip/hip_runtime.h>
#include <hip/hip_bf16.h>
#include <stdint.h>

// ---------------- CDNA5 WMMA types/helpers ----------------
typedef __attribute__((ext_vector_type(16))) __bf16 bf16x16;
typedef __attribute__((ext_vector_type(8)))  float  f32x8;

#define WMMA_BF16(A_, B_, C_) \
  __builtin_amdgcn_wmma_f32_16x16x32_bf16(false, (A_), false, (B_), (short)0, (C_), false, false)

__device__ __forceinline__ unsigned short f2bf(float f) {
  unsigned int u = __float_as_uint(f);
  u += 0x7fffu + ((u >> 16) & 1u);   // round-to-nearest-even
  return (unsigned short)(u >> 16);
}

__device__ __forceinline__ f32x8 zero8() {
  f32x8 z;
#pragma unroll
  for (int i = 0; i < 8; ++i) z[i] = 0.0f;
  return z;
}

// two 16-byte chunks -> one 16-lane bf16 operand
__device__ __forceinline__ bf16x16 ldop(const unsigned short* p0, const unsigned short* p1) {
  union { uint4 q[2]; bf16x16 v; } u;
  u.q[0] = *reinterpret_cast<const uint4*>(p0);
  u.q[1] = *reinterpret_cast<const uint4*>(p1);
  return u.v;
}
__device__ __forceinline__ bf16x16 zop() {
  union { uint4 q[2]; bf16x16 v; } u;
  u.q[0] = make_uint4(0u, 0u, 0u, 0u);
  u.q[1] = make_uint4(0u, 0u, 0u, 0u);
  return u.v;
}

// ---------------- problem constants ----------------
#define BATCH 2
#define TLEN  1024
#define EMB   1024
#define NHEAD 16
#define DHEAD 64
#define JLEN  2304   // 256 + 1024 + 1024
#define MEML  1024
#define CMEML 256
#define QSCALE 0.125f

// ---------------- conversion kernels ----------------
__global__ void k_concat_kv(const float* __restrict__ x, const float* __restrict__ mem,
                            const float* __restrict__ cmem, unsigned short* __restrict__ out) {
  long long idx = (long long)blockIdx.x * 256 + threadIdx.x;
  if (idx >= (long long)BATCH * JLEN * EMB) return;
  int c = (int)(idx % EMB);
  long long rb = idx / EMB;
  int r = (int)(rb % JLEN);
  int b = (int)(rb / JLEN);
  float v;
  if (r < CMEML)              v = cmem[((long long)b * CMEML + r) * EMB + c];
  else if (r < CMEML + MEML)  v = mem[((long long)b * MEML + (r - CMEML)) * EMB + c];
  else                        v = x[((long long)b * TLEN + (r - CMEML - MEML)) * EMB + c];
  out[idx] = f2bf(v);
}

__global__ void k_transpose_cvt(const float* __restrict__ W, unsigned short* __restrict__ WT,
                                int K, int N) {
  long long idx = (long long)blockIdx.x * 256 + threadIdx.x;
  if (idx >= (long long)K * N) return;
  int n = (int)(idx % N);
  long long k = idx / N;
  WT[(long long)n * K + k] = f2bf(W[idx]);
}

__global__ void k_convw(const float* __restrict__ cw, unsigned short* __restrict__ WT) {
  // cw: [O=1024][I=1024][H=4]  ->  WT: [o][h*1024 + i]  (K = 4096)
  long long idx = (long long)blockIdx.x * 256 + threadIdx.x;
  if (idx >= 1024ll * 4096) return;
  int o = (int)(idx / 4096);
  int rem = (int)(idx % 4096);
  int hh = rem / 1024;
  int i = rem % 1024;
  WT[idx] = f2bf(cw[(long long)o * 4096 + (long long)i * 4 + hh]);
}

__global__ void k_cvt(const float* __restrict__ in, unsigned short* __restrict__ out, long long n) {
  long long idx = (long long)blockIdx.x * 256 + threadIdx.x;
  if (idx < n) out[idx] = f2bf(in[idx]);
}

__global__ void k_copyf(const float* __restrict__ in, float* __restrict__ out, long long n) {
  long long idx = (long long)blockIdx.x * 256 + threadIdx.x;
  if (idx < n) out[idx] = in[idx];
}

__global__ void k_zero1(float* p) { if (threadIdx.x == 0 && blockIdx.x == 0) p[0] = 0.0f; }

// ---------------- generic bf16 WMMA GEMM:  C = alpha * A @ BT^T (+bias) ----------------
// A: row-major [M][K] (lda), BT: row-major [N][K] (ldbt).
// mode 0: f32 out to Cf; mode 1: bf16 out to Cb; mode 2: KV split epilogue (K rows / V^T).
// Block 128 threads (4 waves 2x2), block tile 64(M) x 128(N); wave tile 32x64 = 2x4 WMMA
// (8 WMMA per 12 B128 loads per k-step -> 1.5 loads/WMMA operand-reuse).
__global__ __launch_bounds__(128)
void k_gemm(const unsigned short* __restrict__ A, long long strideA, int lda,
            const unsigned short* __restrict__ BT, int ldbt,
            const float* __restrict__ bias, float alpha,
            float* __restrict__ Cf, long long strideC, int ldc,
            unsigned short* __restrict__ Cb,
            unsigned short* __restrict__ Kout, unsigned short* __restrict__ VTout,
            int Kdim, int mode) {
  const int lane = threadIdx.x & 31;
  const int wv = threadIdx.x >> 5;
  const int wm = (wv >> 1) * 32;
  const int wn = (wv & 1) * 64;
  const int m0 = blockIdx.y * 64 + wm;
  const int n0 = blockIdx.x * 128 + wn;
  const int hi = (lane >> 4) & 1;
  const int hi8 = hi * 8;
  const int klo = hi * 16;
  const int lr = lane & 15;

  const unsigned short* Ab = A + (long long)blockIdx.z * strideA;
  const unsigned short* arow0 = Ab + (long long)(m0 + lr) * lda;
  const unsigned short* arow1 = Ab + (long long)(m0 + 16 + lr) * lda;
  const unsigned short* brow0 = BT + (long long)(n0 + lr) * ldbt;
  const unsigned short* brow1 = BT + (long long)(n0 + 16 + lr) * ldbt;
  const unsigned short* brow2 = BT + (long long)(n0 + 32 + lr) * ldbt;
  const unsigned short* brow3 = BT + (long long)(n0 + 48 + lr) * ldbt;

  f32x8 acc[2][4];
#pragma unroll
  for (int mt = 0; mt < 2; ++mt)
#pragma unroll
    for (int nt = 0; nt < 4; ++nt) acc[mt][nt] = zero8();

  for (int k = 0; k < Kdim; k += 32) {
    __builtin_prefetch(arow0 + k + 128, 0, 1);
    __builtin_prefetch(brow0 + k + 128, 0, 1);
    bf16x16 a0 = ldop(arow0 + k + hi8, arow0 + k + 16 + hi8);
    bf16x16 a1 = ldop(arow1 + k + hi8, arow1 + k + 16 + hi8);
    bf16x16 b0 = ldop(brow0 + k + hi8, brow0 + k + 16 + hi8);
    bf16x16 b1 = ldop(brow1 + k + hi8, brow1 + k + 16 + hi8);
    bf16x16 b2 = ldop(brow2 + k + hi8, brow2 + k + 16 + hi8);
    bf16x16 b3 = ldop(brow3 + k + hi8, brow3 + k + 16 + hi8);
    acc[0][0] = WMMA_BF16(a0, b0, acc[0][0]);
    acc[0][1] = WMMA_BF16(a0, b1, acc[0][1]);
    acc[0][2] = WMMA_BF16(a0, b2, acc[0][2]);
    acc[0][3] = WMMA_BF16(a0, b3, acc[0][3]);
    acc[1][0] = WMMA_BF16(a1, b0, acc[1][0]);
    acc[1][1] = WMMA_BF16(a1, b1, acc[1][1]);
    acc[1][2] = WMMA_BF16(a1, b2, acc[1][2]);
    acc[1][3] = WMMA_BF16(a1, b3, acc[1][3]);
  }

#pragma unroll
  for (int mt = 0; mt < 2; ++mt) {
#pragma unroll
    for (int nt = 0; nt < 4; ++nt) {
#pragma unroll
      for (int r = 0; r < 8; ++r) {
        int row = m0 + mt * 16 + r + hi * 8;
        int col = n0 + nt * 16 + lr;
        float v = acc[mt][nt][r] * alpha;
        if (mode == 0) {
          if (bias) v += bias[col];
          Cf[(long long)blockIdx.z * strideC + (long long)row * ldc + col] = v;
        } else if (mode == 1) {
          Cb[(long long)blockIdx.z * strideC + (long long)row * ldc + col] = f2bf(v);
        } else {
          // KV split: rows are (b*2304 + j), cols [0,2048): k then v
          int bb = row / JLEN;
          int j = row - bb * JLEN;
          if (col < EMB) {
            Kout[((long long)bb * JLEN + j) * EMB + col] = f2bf(v);
          } else {
            int d = col - EMB;  // 0..1023 = h*64+dh
            VTout[((long long)bb * EMB + d) * JLEN + j] = f2bf(v);
          }
        }
      }
    }
  }
}

// ---------------- fused flash attention with shifted relative positions ----------------
// grid (T/64, H, B), 128 threads (4 waves, 16 query rows each).
__global__ __launch_bounds__(128)
void k_attn(const unsigned short* __restrict__ Qs,   // [B*T][E], pre-scaled by 1/sqrt(d)
            const unsigned short* __restrict__ Kb,   // [B*J][E]
            const unsigned short* __restrict__ VT,   // [B*E][J]
            const unsigned short* __restrict__ PE,   // [H][J][D]
            unsigned short* __restrict__ Ob) {       // [B*T][E]
  __shared__ float          lsP[4][16][48];
  __shared__ unsigned short lsA[4][16][32];

  const int lane = threadIdx.x & 31;
  const int wv = threadIdx.x >> 5;
  const int b = blockIdx.z;
  const int h = blockIdx.y;
  const int i0 = blockIdx.x * 64;
  const int iw0 = i0 + wv * 16;
  const int hi = (lane >> 4) & 1;
  const int hi8 = hi * 8;
  const int klo = hi * 16;
  const int lr = lane & 15;

  // Q operands (16 rows x 64 K), resident for whole j loop
  const unsigned short* qrow = Qs + ((long long)(b * TLEN + iw0 + lr)) * EMB + h * DHEAD;
  const bf16x16 qa0 = ldop(qrow + hi8, qrow + 16 + hi8);
  const bf16x16 qa1 = ldop(qrow + 32 + hi8, qrow + 48 + hi8);

  const unsigned short* peh = PE + (long long)h * JLEN * DHEAD;

  float rowm[8], rowl[8];
#pragma unroll
  for (int r = 0; r < 8; ++r) { rowm[r] = -3.0e38f; rowl[r] = 0.0f; }
  f32x8 o0 = zero8(), o1 = zero8(), o2 = zero8(), o3 = zero8();

  for (int j0 = 0; j0 < JLEN; j0 += 32) {
    // ---- content scores: S = Qs @ K^T  (16x32) ----
    f32x8 s0 = zero8(), s1 = zero8();
    const unsigned short* kr0 = Kb + ((long long)(b * JLEN + j0 + lr)) * EMB + h * DHEAD;
    const unsigned short* kr1 = kr0 + 16 * EMB;
    {
      bf16x16 kb00 = ldop(kr0 + hi8, kr0 + 16 + hi8);
      bf16x16 kb01 = ldop(kr0 + 32 + hi8, kr0 + 48 + hi8);
      bf16x16 kb10 = ldop(kr1 + hi8, kr1 + 16 + hi8);
      bf16x16 kb11 = ldop(kr1 + 32 + hi8, kr1 + 48 + hi8);
      s0 = WMMA_BF16(qa0, kb00, s0);
      s0 = WMMA_BF16(qa1, kb01, s0);
      s1 = WMMA_BF16(qa0, kb10, s1);
      s1 = WMMA_BF16(qa1, kb11, s1);
    }

    // ---- positional band: P[i, jj] for jj in [jjbase, jjbase+48) ----
    const int jjbase = j0 + (TLEN - 16) - iw0;   // j0 + 1008 - iw0  (>= 0)
    f32x8 p0 = zero8(), p1 = zero8(), p2 = zero8();
#pragma unroll
    for (int pt = 0; pt < 3; ++pt) {
      int prow = jjbase + pt * 16 + lr;
      bf16x16 pb0, pb1;
      if (prow < JLEN) {
        const unsigned short* pr = peh + (long long)prow * DHEAD;
        pb0 = ldop(pr + hi8, pr + 16 + hi8);
        pb1 = ldop(pr + 32 + hi8, pr + 48 + hi8);
      } else {
        pb0 = zop(); pb1 = zop();
      }
      f32x8& pp = (pt == 0) ? p0 : (pt == 1) ? p1 : p2;
      pp = WMMA_BF16(qa0, pb0, pp);
      pp = WMMA_BF16(qa1, pb1, pp);
    }

    // spill P band to LDS (wave-private region), then add shifted values
#pragma unroll
    for (int r = 0; r < 8; ++r) {
      int rr = r + hi * 8;
      lsP[wv][rr][0 * 16 + lr] = p0[r];
      lsP[wv][rr][1 * 16 + lr] = p1[r];
      lsP[wv][rr][2 * 16 + lr] = p2[r];
    }
    __syncthreads();
#pragma unroll
    for (int r = 0; r < 8; ++r) {
      int rr = r + hi * 8;
      s0[r] += lsP[wv][rr][lr + 15 - rr];        // col c = lr,      pcol = c + 15 - rr
      s1[r] += lsP[wv][rr][16 + lr + 15 - rr];   // col c = 16 + lr
    }

    // ---- online softmax (row data lives across a 16-lane half) ----
#pragma unroll
    for (int r = 0; r < 8; ++r) {
      float t = fmaxf(s0[r], s1[r]);
      t = fmaxf(t, __shfl_xor(t, 1));
      t = fmaxf(t, __shfl_xor(t, 2));
      t = fmaxf(t, __shfl_xor(t, 4));
      t = fmaxf(t, __shfl_xor(t, 8));
      float mn = fmaxf(rowm[r], t);
      float corr = __expf(rowm[r] - mn);
      rowm[r] = mn;
      float e0 = __expf(s0[r] - mn);
      float e1 = __expf(s1[r] - mn);
      s0[r] = e0; s1[r] = e1;
      float su = e0 + e1;
      su += __shfl_xor(su, 1);
      su += __shfl_xor(su, 2);
      su += __shfl_xor(su, 4);
      su += __shfl_xor(su, 8);
      rowl[r] = rowl[r] * corr + su;
      o0[r] *= corr; o1[r] *= corr; o2[r] *= corr; o3[r] *= corr;
    }

    // ---- probs -> bf16 A operand via LDS ----
#pragma unroll
    for (int r = 0; r < 8; ++r) {
      int rr = r + hi * 8;
      lsA[wv][rr][lr] = f2bf(s0[r]);
      lsA[wv][rr][16 + lr] = f2bf(s1[r]);
    }
    __syncthreads();
    bf16x16 pa = ldop(&lsA[wv][lr][hi8], &lsA[wv][lr][16 + hi8]);

    // ---- O += P @ V  (V stored transposed: [d][j]) ----
#pragma unroll
    for (int nt = 0; nt < 4; ++nt) {
      const unsigned short* vrow =
          VT + ((long long)(b * EMB + h * DHEAD + nt * 16 + lr)) * JLEN + j0;
      bf16x16 vb = ldop(vrow + hi8, vrow + 16 + hi8);
      f32x8& oo = (nt == 0) ? o0 : (nt == 1) ? o1 : (nt == 2) ? o2 : o3;
      oo = WMMA_BF16(pa, vb, oo);
    }
  }

  // normalize + store bf16 [b, i, h*64 + d]
#pragma unroll
  for (int nt = 0; nt < 4; ++nt) {
    f32x8& oo = (nt == 0) ? o0 : (nt == 1) ? o1 : (nt == 2) ? o2 : o3;
#pragma unroll
    for (int r = 0; r < 8; ++r) {
      int rr = r + hi * 8;
      float v = oo[r] / rowl[r];
      Ob[((long long)(b * TLEN + iw0 + rr)) * EMB + h * DHEAD + nt * 16 + lr] = f2bf(v);
    }
  }
}

// ---------------- host launcher ----------------
extern "C" void kernel_launch(void* const* d_in, const int* in_sizes, int n_in,
                              void* d_out, int out_size, void* d_ws, size_t ws_size,
                              hipStream_t stream) {
  const float* x      = (const float*)d_in[0];
  const float* mem    = (const float*)d_in[1];
  const float* cmem   = (const float*)d_in[2];
  const float* pos    = (const float*)d_in[3];
  // d_in[4] = input_mask (all true -> unused)
  const float* Wq     = (const float*)d_in[5];
  const float* Wkv    = (const float*)d_in[6];
  const float* Wo     = (const float*)d_in[7];
  const float* bo     = (const float*)d_in[8];
  const float* convw  = (const float*)d_in[9];
  const float* convb  = (const float*)d_in[10];
  float* out = (float*)d_out;

  char* ws = (char*)d_ws;
  size_t o = 0;
  unsigned short* kvinb = (unsigned short*)(ws + o); o += 2ull * JLEN * EMB * 2;   // 9.4 MB
  unsigned short* WqT   = (unsigned short*)(ws + o); o += 1024ull * 1024 * 2;
  unsigned short* WkvT  = (unsigned short*)(ws + o); o += 2048ull * 1024 * 2;
  unsigned short* WoT   = (unsigned short*)(ws + o); o += 1024ull * 1024 * 2;
  unsigned short* W2T   = (unsigned short*)(ws + o); o += 1024ull * 4096 * 2;
  unsigned short* peb   = (unsigned short*)(ws + o); o += (size_t)NHEAD * JLEN * DHEAD * 2;
  unsigned short* Qs    = (unsigned short*)(ws + o); o += 2048ull * 1024 * 2;
  unsigned short* Kb    = (unsigned short*)(ws + o); o += 2ull * JLEN * EMB * 2;
  unsigned short* VT    = (unsigned short*)(ws + o); o += 2ull * EMB * JLEN * 2;
  unsigned short* Ob    = (unsigned short*)(ws + o); o += 2048ull * 1024 * 2;      // total ~58 MB

  // --- conversions ---
  {
    long long n = 2ll * JLEN * EMB;
    k_concat_kv<<<(unsigned)((n + 255) / 256), 256, 0, stream>>>(x, mem, cmem, kvinb);
  }
  k_transpose_cvt<<<(1024u * 1024u + 255) / 256, 256, 0, stream>>>(Wq, WqT, 1024, 1024);
  k_transpose_cvt<<<(1024u * 2048u + 255) / 256, 256, 0, stream>>>(Wkv, WkvT, 1024, 2048);
  k_transpose_cvt<<<(1024u * 1024u + 255) / 256, 256, 0, stream>>>(Wo, WoT, 1024, 1024);
  k_convw<<<(1024u * 4096u + 255) / 256, 256, 0, stream>>>(convw, W2T);
  {
    long long n = (long long)NHEAD * JLEN * DHEAD;
    k_cvt<<<(unsigned)((n + 255) / 256), 256, 0, stream>>>(pos, peb, n);
  }

  // --- Q = x @ Wq, scaled, bf16 out.  x lives at rows [1280,2304) of each kv slab ---
  k_gemm<<<dim3(8, 16, 2), 128, 0, stream>>>(
      kvinb + 1280ll * EMB, 2304ll * EMB, EMB, WqT, 1024, nullptr, QSCALE,
      nullptr, 1024ll * 1024, 1024, Qs, nullptr, nullptr, 1024, 1);

  // --- KV = kv_input @ Wkv, split epilogue (K row-major, V transposed) ---
  k_gemm<<<dim3(16, 72, 1), 128, 0, stream>>>(
      kvinb, 0, EMB, WkvT, 1024, nullptr, 1.0f,
      nullptr, 0, 0, nullptr, Kb, VT, 1024, 2);

  // --- fused attention ---
  k_attn<<<dim3(TLEN / 64, NHEAD, BATCH), 128, 0, stream>>>(Qs, Kb, VT, peb, Ob);

  // --- logits = O @ Wo + bo -> d_out[0 : 2*1024*1024) ---
  k_gemm<<<dim3(8, 32, 1), 128, 0, stream>>>(
      Ob, 0, EMB, WoT, 1024, bo, 1.0f,
      out, 0, 1024, nullptr, nullptr, nullptr, 1024, 0);

  // --- new_mem = x -> d_out[2M : 4M) ---
  k_copyf<<<(2u * 1024 * 1024 + 255) / 256, 256, 0, stream>>>(
      x, out + 2ll * 1024 * 1024, 2ll * 1024 * 1024);

  // --- new_cmem = conv(mem) + conv_b -> d_out[4M : 4M+512K). mem = rows [256,1280) of kv slab,
  //     viewed per batch as [256][4096] with lda 4096; weights pre-arranged as [1024][4096]^T ---
  k_gemm<<<dim3(8, 4, 2), 128, 0, stream>>>(
      kvinb + 256ll * EMB, 2304ll * EMB, 4096, W2T, 4096, convb, 1.0f,
      out + 4ll * 1024 * 1024, 256ll * 1024, 1024, nullptr, nullptr, nullptr, 4096, 0);

  // --- aux_loss = 0 ---
  k_zero1<<<1, 1, 0, stream>>>(out + 4ll * 1024 * 1024 + 2ll * 256 * 1024);
}
